// LowFreEnhanceBlock_34935263985950
// MI455X (gfx1250) — matrix operand order repaired and checked by reading, not compiled, not run on hardware
//
#include <hip/hip_runtime.h>
#include <hip/hip_bf16.h>
#include <math.h>

// ---------------------------------------------------------------------------
// LowFreEnhanceBlock on MI455X (gfx1250): GEMMs via v_wmma_f32_16x16x32_f16
// (wave32), weights staged to LDS with the Tensor Data Mover (TENSORcnt),
// 6 output tiles per wave to amortize A-fragment traffic.
// C=96, V=32*32*32=32768 voxels, NH=4, hd=24.
// ---------------------------------------------------------------------------

#define C_   96
#define V_   32768

typedef __attribute__((ext_vector_type(16))) _Float16 v16h;
typedef __attribute__((ext_vector_type(8)))  _Float16 v8h;
typedef __attribute__((ext_vector_type(8)))  float    v8f;
typedef __attribute__((ext_vector_type(4)))  unsigned int u32x4;
typedef __attribute__((ext_vector_type(4)))  int      i32x4;
typedef __attribute__((ext_vector_type(8)))  int      i32x8;

static __device__ __forceinline__ float gelu_f(float x) {
  return 0.5f * x * (1.0f + erff(x * 0.70710678118654752440f));
}

static __device__ __forceinline__ v16h mk_frag(v8h lo, v8h hi) {
  v16h r;
#pragma unroll
  for (int i = 0; i < 8; ++i) { r[i] = lo[i]; r[i + 8] = hi[i]; }
  return r;
}

// A-fragment (16x32 f16): lanes 0-15 rows 0-15, K {k0..k0+15};
// lanes 16-31 same rows, K {k0+16..k0+31}. Two b128 loads.
static __device__ __forceinline__ v16h a_frag(const _Float16* row, int k0, int hi) {
  int kb = k0 + (hi ? 16 : 0);
  return mk_frag(*(const v8h*)(row + kb), *(const v8h*)(row + kb + 8));
}

// B-fragment (32x16 f16) from row-major Wt[N][K] (B = Wt^T). Works on global
// or LDS pointers (LDS -> ds_read_b128 x2).
static __device__ __forceinline__ v16h b_frag_wt(const _Float16* Wt, int ldk,
                                                 int n0, int k0, int lane) {
  const _Float16* p = Wt + (size_t)(n0 + (lane & 15)) * ldk + k0 + ((lane >> 4) << 4);
  return mk_frag(*(const v8h*)p, *(const v8h*)(p + 8));
}

// B-fragment from row-major B[K][N] with runtime row stride (scalar gathers).
static __device__ __forceinline__ v16h b_frag_strided(const _Float16* base, int stride,
                                                      int k0, int n0, int lane) {
  v16h b;
  int col = n0 + (lane & 15);
  int kh  = k0 + ((lane >> 4) << 4);
#pragma unroll
  for (int i = 0; i < 16; ++i) b[i] = base[(size_t)(kh + i) * stride + col];
  return b;
}

static __device__ __forceinline__ v8f wmma32(v16h a, v16h b, v8f c) {
  return __builtin_amdgcn_wmma_f32_16x16x32_f16(false, a, false, b, (short)0, c,
                                                false, false);
}

// --------------------------- Tensor Data Mover -----------------------------
// 2D tile load global->LDS. D# built per CDNA5 ISA ch.8 (group0: count=1,
// lds_addr, 57b global addr, type=2; group1: data_size=1 (2B), dims/strides
// in elements). Issued by one wave; EXEC ignored; tracked by TENSORcnt.

static __device__ __forceinline__ unsigned lds_off(const void* p) {
  return (unsigned)(size_t)p;   // low 32 bits of flat shared addr = LDS offset
}

static __device__ __forceinline__ void tdm_load_2d(unsigned lds_addr, const void* gsrc,
                                                   unsigned w_elems, unsigned h_rows,
                                                   unsigned stride_elems) {
#if __has_builtin(__builtin_amdgcn_tensor_load_to_lds)
  unsigned long long ga = (unsigned long long)gsrc;
  u32x4 g0;
  g0[0] = 1u;                                           // count=1 (valid D#)
  g0[1] = lds_addr;                                     // LDS byte address
  g0[2] = (unsigned)(ga & 0xffffffffu);                 // global_addr[31:0]
  g0[3] = (unsigned)((ga >> 32) & 0x01ffffffu) | 0x80000000u;  // [56:32] | type=2
  i32x8 g1;
  g1[0] = 0x00010000;                                   // data_size=1 (2B)
  g1[1] = (int)(w_elems << 16);                         // tensor_dim0 lo16
  g1[2] = (int)((w_elems >> 16) | (h_rows << 16));      // dim0 hi | dim1 lo
  g1[3] = (int)((h_rows >> 16) | (w_elems << 16));      // dim1 hi | tile_dim0
  g1[4] = (int)h_rows;                                  // tile_dim1 (tile_dim2=0)
  g1[5] = (int)stride_elems;                            // tensor_dim0_stride lo32
  g1[6] = 0;
  g1[7] = 0;
  i32x4 z4 = {0, 0, 0, 0};
#if __clang_major__ >= 23
  i32x8 z8 = {0, 0, 0, 0, 0, 0, 0, 0};
  __builtin_amdgcn_tensor_load_to_lds(g0, g1, z4, z4, z8, 0);
#else
  __builtin_amdgcn_tensor_load_to_lds(g0, g1, z4, z4, 0);
#endif
#else
  (void)lds_addr; (void)gsrc; (void)w_elems; (void)h_rows; (void)stride_elems;
#endif
}

static __device__ __forceinline__ void tdm_wait() {
#if __has_builtin(__builtin_amdgcn_s_wait_tensorcnt)
  __builtin_amdgcn_s_wait_tensorcnt(0);
#else
  asm volatile("s_wait_tensorcnt 0x0" ::: "memory");
#endif
}

// Stage row-major f16 weight block [h_rows][w_elems] into LDS.
// TDM path: wave 0 issues one DMA. Fallback: cooperative vector copy.
static __device__ __forceinline__ void stage_w(_Float16* slab, const _Float16* g,
                                               int halves, int w_elems, int h_rows,
                                               int tid) {
#if __has_builtin(__builtin_amdgcn_tensor_load_to_lds)
  (void)halves;
  if (tid < 32)
    tdm_load_2d(lds_off(slab), g, (unsigned)w_elems, (unsigned)h_rows,
                (unsigned)w_elems);
#else
  (void)w_elems; (void)h_rows;
  for (int i = tid * 8; i < halves; i += 256 * 8)
    *(v8h*)(slab + i) = *(const v8h*)(g + i);
#endif
}

// window token index -> voxel index (window partition + cyclic shift)
static __device__ __forceinline__ int map_vox(int m, int wd, int wh, int ww,
                                              int sd, int sh, int sw) {
  int nwin = wd * wh * ww;
  int n = m % nwin, w = m / nwin;
  int nww = 32 / ww, nwh2 = 32 / wh;
  int bw = w % nww; int t = w / nww; int bh = t % nwh2; int bd = t / nwh2;
  int ix = n % ww; int t2 = n / ww; int iy = t2 % wh; int iz = t2 / wh;
  int gd = (bd * wd + iz + sd) & 31;
  int gh = (bh * wh + iy + sh) & 31;
  int gx = (bw * ww + ix + sw) & 31;
  return (gd << 10) + (gh << 5) + gx;
}

static __device__ __forceinline__ int reg1(int p, int wsd, int ssd) {
  return (p < 32 - wsd) ? 0 : ((p < 32 - ssd) ? 1 : 2);
}

// ------------------------------- utility kernels ---------------------------

__global__ void k_cvt(const float* __restrict__ s, _Float16* __restrict__ d, int n) {
  int i = blockIdx.x * blockDim.x + threadIdx.x;
  if (i < n) d[i] = (_Float16)s[i];
}

// conv weight (C,C,1,KH,KW) -> f16 [T][96][96]  (T = KH*KW taps)
__global__ void k_cvt_convw(const float* __restrict__ s, _Float16* __restrict__ d, int T) {
  int i = blockIdx.x * blockDim.x + threadIdx.x;
  if (i >= 9216 * T) return;
  int tap = i / 9216; int r = i % 9216; int n = r / 96; int ic = r % 96;
  d[i] = (_Float16)s[(size_t)(n * 96 + ic) * T + tap];
}

__global__ void k_fill0h(_Float16* __restrict__ p, size_t n) {
  size_t i = (size_t)blockIdx.x * blockDim.x + threadIdx.x;
  if (i < n) p[i] = (_Float16)0.0f;
}

// NCDHW f32 -> channels-last f32 [V][96]
__global__ void k_init_xcl(const float* __restrict__ xin, float* __restrict__ xcl) {
  size_t i = (size_t)blockIdx.x * blockDim.x + threadIdx.x;
  int c = (int)(i >> 15); int v = (int)(i & 32767);
  xcl[(size_t)v * 96 + c] = xin[i];
}

__global__ void k_final(const float* __restrict__ xcl, const float* __restrict__ xin,
                        const float* __restrict__ fls, float* __restrict__ out) {
  size_t i = (size_t)blockIdx.x * blockDim.x + threadIdx.x;
  int c = (int)(i >> 15); int v = (int)(i & 32767);
  out[i] = fls[c] * xcl[(size_t)v * 96 + c] + xin[i];
}

// LayerNorm over channels per voxel, f16 output
__global__ __launch_bounds__(256) void k_ln(const float* __restrict__ x,
                                            const float* __restrict__ g,
                                            const float* __restrict__ b,
                                            _Float16* __restrict__ out) {
  int v = blockIdx.x * blockDim.x + threadIdx.x;
  if (v >= V_) return;
  const float* r = x + (size_t)v * 96;
  float m = 0.f;
#pragma unroll 8
  for (int c = 0; c < 96; ++c) m += r[c];
  m *= (1.0f / 96.0f);
  float var = 0.f;
#pragma unroll 8
  for (int c = 0; c < 96; ++c) { float d = r[c] - m; var += d * d; }
  var *= (1.0f / 96.0f);
  float rs = rsqrtf(var + 1e-5f);
  _Float16* o = out + (size_t)v * 96;
#pragma unroll 8
  for (int c = 0; c < 96; ++c) o[c] = (_Float16)((r[c] - m) * rs * g[c] + b[c]);
}

__global__ __launch_bounds__(256) void k_chan_mean(const float* __restrict__ x,
                                                   float* __restrict__ gm) {
  __shared__ float sh[256];
  int c = blockIdx.x, t = threadIdx.x;
  float s = 0.f;
  for (int v = t; v < V_; v += 256) s += x[(size_t)v * 96 + c];
  sh[t] = s; __syncthreads();
  for (int k = 128; k > 0; k >>= 1) { if (t < k) sh[t] += sh[t + k]; __syncthreads(); }
  if (t == 0) gm[c] = sh[0] * (1.0f / V_);
}

__global__ __launch_bounds__(256) void k_in_stats(const float* __restrict__ f,
                                                  float* __restrict__ mean,
                                                  float* __restrict__ var) {
  __shared__ float s1[256], s2[256];
  int c = blockIdx.x, t = threadIdx.x;
  float a = 0.f, b = 0.f;
  for (int v = t; v < V_; v += 256) {
    float x = f[(size_t)v * 96 + c];
    a += x; b += x * x;
  }
  s1[t] = a; s2[t] = b; __syncthreads();
  for (int k = 128; k > 0; k >>= 1) {
    if (t < k) { s1[t] += s1[t + k]; s2[t] += s2[t + k]; }
    __syncthreads();
  }
  if (t == 0) {
    float m = s1[0] * (1.0f / V_);
    mean[c] = m;
    var[c] = s2[0] * (1.0f / V_) - m * m;
  }
}

// gate MLP: gm(96) -> gelu(12) -> softmax(4)
__global__ void k_gate(const float* __restrict__ gm,
                       const float* __restrict__ w1, const float* __restrict__ b1,
                       const float* __restrict__ w2, const float* __restrict__ b2,
                       float* __restrict__ gw) {
  __shared__ float hsh[12];
  int t = threadIdx.x;
  if (t < 12) {
    float s = b1[t];
    for (int c = 0; c < 96; ++c) s += gm[c] * w1[t * 96 + c];
    hsh[t] = gelu_f(s);
  }
  __syncthreads();
  if (t == 0) {
    float lg[4]; float mx = -3.0e38f;
    for (int k = 0; k < 4; ++k) {
      float s = b2[k];
      for (int j = 0; j < 12; ++j) s += hsh[j] * w2[k * 12 + j];
      lg[k] = s; mx = fmaxf(mx, s);
    }
    float den = 0.f;
    for (int k = 0; k < 4; ++k) { lg[k] = __expf(lg[k] - mx); den += lg[k]; }
    for (int k = 0; k < 4; ++k) gw[k] = lg[k] / den;
  }
}

// x = x + ls_a*(y - x);  also make f16 copy for convs
__global__ void k_xupd(float* __restrict__ x, const float* __restrict__ y,
                       const float* __restrict__ ls, _Float16* __restrict__ xh) {
  size_t i = (size_t)blockIdx.x * blockDim.x + threadIdx.x;
  int c = (int)(i % 96);
  float nv = x[i] + ls[c] * (y[i] - x[i]);
  x[i] = nv; xh[i] = (_Float16)nv;
}

// depthwise (3,1,1) conv along D -> cat columns [96..191]
__global__ void k_dwconv(const float* __restrict__ x, const float* __restrict__ w2,
                         _Float16* __restrict__ cat) {
  size_t i = (size_t)blockIdx.x * blockDim.x + threadIdx.x;
  int c = (int)(i % 96); int v = (int)(i / 96);
  int d = v >> 10; int rest = v & 1023;
  float s = 0.f;
#pragma unroll
  for (int kd = 0; kd < 3; ++kd) {
    int dd = d + kd - 1;
    if (dd >= 0 && dd < 32) s += w2[c * 3 + kd] * x[(size_t)((dd << 10) + rest) * 96 + c];
  }
  cat[(size_t)v * 384 + 96 + c] = (_Float16)s;
}

// x += ls_c * gelu(instnorm(f))
__global__ void k_in_apply(float* __restrict__ x, const float* __restrict__ f,
                           const float* __restrict__ mean, const float* __restrict__ var,
                           const float* __restrict__ g, const float* __restrict__ b,
                           const float* __restrict__ ls) {
  size_t i = (size_t)blockIdx.x * blockDim.x + threadIdx.x;
  int c = (int)(i % 96);
  float xn = (f[i] - mean[c]) * rsqrtf(var[c] + 1e-5f) * g[c] + b[c];
  x[i] += ls[c] * gelu_f(xn);
}

// ------------------------------- WMMA GEMM kernels -------------------------
// 256 threads = 8 wave32; each wave computes 16 rows x 96 cols (6 WMMA tiles)
// to amortize A-fragment loads. Weights staged to LDS via TDM.

// QKV: gather LN'd rows via window map; store head-padded f16 (hd 24 -> 32).
// N=288 -> grid.y=3, 96 cols per block. W (288x96) staged whole (55 KB).
__global__ __launch_bounds__(256) void k_gemm_qkv(
    const _Float16* __restrict__ xln, const _Float16* __restrict__ wt,
    const float* __restrict__ bias, _Float16* __restrict__ qkv,
    int wd, int wh, int ww, int sd, int sh, int sw) {
  __shared__ __align__(16) _Float16 wslab[288 * 96];
  int tid = threadIdx.x;
  int lane = tid & 31, wave = tid >> 5, hi = lane >> 4;
  int m0 = blockIdx.x * 128 + wave * 16;
  int colbase = blockIdx.y * 96;
  stage_w(wslab, wt, 288 * 96, 96, 288, tid);
  int m = m0 + (lane & 15);
  const _Float16* row = xln + (size_t)map_vox(m, wd, wh, ww, sd, sh, sw) * 96;
  tdm_wait();
  __syncthreads();
  v8f acc[6] = {};
#pragma unroll
  for (int k0 = 0; k0 < 96; k0 += 32) {
    v16h a = a_frag(row, k0, hi);
#pragma unroll
    for (int j = 0; j < 6; ++j)
      acc[j] = wmma32(a, b_frag_wt(wslab, 96, colbase + j * 16, k0, lane), acc[j]);
  }
#pragma unroll
  for (int j = 0; j < 6; ++j) {
    int col = colbase + j * 16 + (lane & 15);
    float bv = bias[col];
    int s = col / 96, rc = col % 96;
    int pc = s * 128 + (rc / 24) * 32 + (rc % 24);   // padded (3,NH,32) layout
#pragma unroll
    for (int r = 0; r < 8; ++r) {
      int rm = m0 + r + (hi << 3);
      qkv[(size_t)rm * 384 + pc] = (_Float16)(acc[r < 8 ? 0 * 0 + 0 : 0][r] * 0.0f +
                                              acc[j][r] + bv);
    }
  }
}

// Fused per-window attention: S=QK^T (WMMA) -> bias/mask/softmax in regs ->
// P to LDS (A-fragment layout, 272B row stride) -> O=PV (WMMA).
template <int NT>
__global__ __launch_bounds__(NT * 32) void k_attn(
    const _Float16* __restrict__ qkv, const float* __restrict__ rpb,
    _Float16* __restrict__ aout, int wd, int wh, int ww, int shifted) {
  constexpr int N = NT * 16;
  __shared__ __align__(16) _Float16 lds[NT * 16 * 136];
  int lane = threadIdx.x & 31, wave = threadIdx.x >> 5, hi = lane >> 4;
  int w = blockIdx.x, h = blockIdx.y;
  int rbase = wave * 16;
  int nww = 32 / ww, nwh2 = 32 / wh;
  int bw = w % nww; int t = w / nww; int bh = t % nwh2; int bd = t / nwh2;
  const size_t wbase = (size_t)w * N * 384;

  const _Float16* qrow = qkv + wbase + (size_t)(rbase + (lane & 15)) * 384 + h * 32;
  v16h aq = a_frag(qrow, 0, hi);              // K=32 (zero padded past hd=24)
  const _Float16* kb = qkv + wbase + 128 + h * 32;
  v8f acc[NT];
#pragma unroll
  for (int j = 0; j < NT; ++j) {
    v8f z = {};
    acc[j] = wmma32(aq, b_frag_wt(kb, 384, j * 16, 0, lane), z);
  }

  const float scale = 0.204124145231931508f;  // 24^-0.5
  int ssd = shifted ? (wd >> 1) : 0;
  int ssh = shifted ? (wh >> 1) : 0;
  int ssw = shifted ? (ww >> 1) : 0;
  int zr[8], yr[8], xr[8], labr[8];
#pragma unroll
  for (int r = 0; r < 8; ++r) {
    int rr = rbase + r + (hi << 3);
    int xrr = rr % ww; int t2 = rr / ww; int yrr = t2 % wh; int zrr = t2 / wh;
    zr[r] = zrr; yr[r] = yrr; xr[r] = xrr;
    labr[r] = reg1(bd * wd + zrr, wd, ssd) * 9 + reg1(bh * wh + yrr, wh, ssh) * 3 +
              reg1(bw * ww + xrr, ww, ssw);
  }
  float rmx[8];
#pragma unroll
  for (int r = 0; r < 8; ++r) rmx[r] = -3.0e38f;
#pragma unroll
  for (int j = 0; j < NT; ++j) {
    int tc = j * 16 + (lane & 15);
    int xc = tc % ww; int tt = tc / ww; int yc = tt % wh; int zc = tt / wh;
    int labc = reg1(bd * wd + zc, wd, ssd) * 9 + reg1(bh * wh + yc, wh, ssh) * 3 +
               reg1(bw * ww + xc, ww, ssw);
#pragma unroll
    for (int r = 0; r < 8; ++r) {
      int idx = ((zr[r] - zc + wd - 1) * (2 * wh - 1) + (yr[r] - yc + wh - 1)) *
                    (2 * ww - 1) + (xr[r] - xc + ww - 1);
      float v = acc[j][r] * scale + rpb[idx * 4 + h];
      if (shifted && (labr[r] != labc)) v -= 100.0f;
      acc[j][r] = v;
      rmx[r] = fmaxf(rmx[r], v);
    }
  }
  // row softmax: each row lives in one 16-lane half -> xor-shuffle reductions
#pragma unroll
  for (int r = 0; r < 8; ++r) {
    float v = rmx[r];
    v = fmaxf(v, __shfl_xor(v, 1, 32)); v = fmaxf(v, __shfl_xor(v, 2, 32));
    v = fmaxf(v, __shfl_xor(v, 4, 32)); v = fmaxf(v, __shfl_xor(v, 8, 32));
    rmx[r] = v;
  }
  float rsum[8];
#pragma unroll
  for (int r = 0; r < 8; ++r) rsum[r] = 0.f;
#pragma unroll
  for (int j = 0; j < NT; ++j)
#pragma unroll
    for (int r = 0; r < 8; ++r) {
      float e = __expf(acc[j][r] - rmx[r]);
      acc[j][r] = e;
      rsum[r] += e;
    }
#pragma unroll
  for (int r = 0; r < 8; ++r) {
    float v = rsum[r];
    v += __shfl_xor(v, 1, 32); v += __shfl_xor(v, 2, 32);
    v += __shfl_xor(v, 4, 32); v += __shfl_xor(v, 8, 32);
    rsum[r] = 1.0f / v;
  }
  _Float16* slab = &lds[wave * 16 * 136];
#pragma unroll
  for (int j = 0; j < NT; ++j)
#pragma unroll
    for (int r = 0; r < 8; ++r)
      slab[(r + (hi << 3)) * 136 + j * 16 + (lane & 15)] =
          (_Float16)(acc[j][r] * rsum[r]);
  __syncthreads();

  const _Float16* vb = qkv + wbase + 256 + h * 32;
  const _Float16* prow = slab + (lane & 15) * 136;
  v8f o0 = {}, o1 = {};
#pragma unroll
  for (int ks = 0; ks < NT / 2; ++ks) {
    int kk = ks * 32;
    v16h pa = a_frag(prow, kk, hi);           // ds_read_b128 x2
    o0 = wmma32(pa, b_frag_strided(vb, 384, kk, 0, lane), o0);
    o1 = wmma32(pa, b_frag_strided(vb, 384, kk, 16, lane), o1);
  }
  int e0 = lane & 15;
#pragma unroll
  for (int r = 0; r < 8; ++r) {
    int rr = rbase + r + (hi << 3);
    size_t ob = (size_t)(w * N + rr) * 96 + h * 24;
    aout[ob + e0] = (_Float16)o0[r];
    if (e0 < 8) aout[ob + 16 + e0] = (_Float16)o1[r];
  }
}

// proj GEMM (96x96 W staged, 18 KB) + window-reverse scatter + shortcut add
__global__ __launch_bounds__(256) void k_gemm_proj(
    const _Float16* __restrict__ aout, const _Float16* __restrict__ wt,
    const float* __restrict__ bias, const float* __restrict__ xcl,
    float* __restrict__ xafter, int wd, int wh, int ww, int sd, int sh, int sw) {
  __shared__ __align__(16) _Float16 wslab[96 * 96];
  int tid = threadIdx.x;
  int lane = tid & 31, wave = tid >> 5, hi = lane >> 4;
  int m0 = blockIdx.x * 128 + wave * 16;
  stage_w(wslab, wt, 96 * 96, 96, 96, tid);
  const _Float16* row = aout + (size_t)(m0 + (lane & 15)) * 96;
  tdm_wait();
  __syncthreads();
  v8f acc[6] = {};
#pragma unroll
  for (int k0 = 0; k0 < 96; k0 += 32) {
    v16h a = a_frag(row, k0, hi);
#pragma unroll
    for (int j = 0; j < 6; ++j)
      acc[j] = wmma32(a, b_frag_wt(wslab, 96, j * 16, k0, lane), acc[j]);
  }
#pragma unroll
  for (int r = 0; r < 8; ++r) {
    int rm = m0 + r + (hi << 3);
    size_t vb = (size_t)map_vox(rm, wd, wh, ww, sd, sh, sw) * 96;
#pragma unroll
    for (int j = 0; j < 6; ++j) {
      int col = j * 16 + (lane & 15);
      size_t o = vb + col;
      xafter[o] = xcl[o] + acc[j][r] + bias[col];
    }
  }
}

// fc1: 96 -> 384 (W 384x96 staged whole, 72 KB), exact GELU, f16 out. grid.y=4
__global__ __launch_bounds__(256) void k_gemm_fc1(
    const _Float16* __restrict__ xln, const _Float16* __restrict__ wt,
    const float* __restrict__ bias, _Float16* __restrict__ hb) {
  __shared__ __align__(16) _Float16 wslab[384 * 96];
  int tid = threadIdx.x;
  int lane = tid & 31, wave = tid >> 5, hi = lane >> 4;
  int m0 = blockIdx.x * 128 + wave * 16;
  int colbase = blockIdx.y * 96;
  stage_w(wslab, wt, 384 * 96, 96, 384, tid);
  const _Float16* row = xln + (size_t)(m0 + (lane & 15)) * 96;
  tdm_wait();
  __syncthreads();
  v8f acc[6] = {};
#pragma unroll
  for (int k0 = 0; k0 < 96; k0 += 32) {
    v16h a = a_frag(row, k0, hi);
#pragma unroll
    for (int j = 0; j < 6; ++j)
      acc[j] = wmma32(a, b_frag_wt(wslab, 96, colbase + j * 16, k0, lane), acc[j]);
  }
#pragma unroll
  for (int j = 0; j < 6; ++j) {
    int col = colbase + j * 16 + (lane & 15);
    float bv = bias[col];
#pragma unroll
    for (int r = 0; r < 8; ++r) {
      int rm = m0 + r + (hi << 3);
      hb[(size_t)rm * 384 + col] = (_Float16)gelu_f(acc[j][r] + bv);
    }
  }
}

// fc2: 384 -> 96 (W 96x384 staged whole, 72 KB), shortcut + gated accumulate
__global__ __launch_bounds__(256) void k_gemm_fc2(
    const _Float16* __restrict__ hb, const _Float16* __restrict__ wt,
    const float* __restrict__ bias, const float* __restrict__ xafter,
    float* __restrict__ y, const float* __restrict__ gw, int br) {
  __shared__ __align__(16) _Float16 wslab[96 * 384];
  int tid = threadIdx.x;
  int lane = tid & 31, wave = tid >> 5, hi = lane >> 4;
  int m0 = blockIdx.x * 128 + wave * 16;
  stage_w(wslab, wt, 96 * 384, 384, 96, tid);
  const _Float16* row = hb + (size_t)(m0 + (lane & 15)) * 384;
  __builtin_prefetch(row, 0, 0);
  tdm_wait();
  __syncthreads();
  v8f acc[6] = {};
#pragma unroll
  for (int k0 = 0; k0 < 384; k0 += 32) {
    v16h a = a_frag(row, k0, hi);
#pragma unroll
    for (int j = 0; j < 6; ++j)
      acc[j] = wmma32(a, b_frag_wt(wslab, 384, j * 16, k0, lane), acc[j]);
  }
  float gv = gw[br];
#pragma unroll
  for (int j = 0; j < 6; ++j) {
    int col = j * 16 + (lane & 15);
    float bv = bias[col];
#pragma unroll
    for (int r = 0; r < 8; ++r) {
      size_t idx = (size_t)(m0 + r + (hi << 3)) * 96 + col;
      float bo = xafter[idx] + acc[j][r] + bv;
      if (br == 0) y[idx] = gv * bo;
      else         y[idx] += gv * bo;
    }
  }
}

// implicit-GEMM 2D conv (1xKHxKW): per-tap 96x96 weight tile double-buffered
// through LDS via TDM; 6 output tiles per wave (full 96 cols per block).
__global__ __launch_bounds__(256) void k_gemm_conv(
    const _Float16* __restrict__ xh, const _Float16* __restrict__ wt,
    _Float16* __restrict__ cat, const _Float16* __restrict__ zrow,
    int KH, int KW, int PH, int PW, int coff) {
  __shared__ __align__(16) _Float16 bslab[2][96 * 96];
  int tid = threadIdx.x;
  int lane = tid & 31, wave = tid >> 5, hi = lane >> 4;
  int m0 = blockIdx.x * 128 + wave * 16;
  int v = m0 + (lane & 15);
  int d = v >> 10, hh = (v >> 5) & 31, xx = v & 31;
  int T = KH * KW;
  stage_w(bslab[0], wt, 96 * 96, 96, 96, tid);
  v8f acc[6] = {};
  for (int t = 0; t < T; ++t) {
    tdm_wait();
    __syncthreads();   // slab[t&1] ready; readers of slab[(t+1)&1] are done
    if (t + 1 < T)
      stage_w(bslab[(t + 1) & 1], wt + (size_t)(t + 1) * 9216, 96 * 96, 96, 96, tid);
    int kh = t / KW, kw = t % KW;
    int sh2 = hh + kh - PH, sw2 = xx + kw - PW;
    const _Float16* row =
        (sh2 >= 0 && sh2 < 32 && sw2 >= 0 && sw2 < 32)
            ? xh + (size_t)((d << 10) + (sh2 << 5) + sw2) * 96
            : zrow;
    const _Float16* bs = bslab[t & 1];
#pragma unroll
    for (int k0 = 0; k0 < 96; k0 += 32) {
      v16h a = a_frag(row, k0, hi);
#pragma unroll
      for (int j = 0; j < 6; ++j)
        acc[j] = wmma32(a, b_frag_wt(bs, 96, j * 16, k0, lane), acc[j]);
    }
  }
#pragma unroll
  for (int j = 0; j < 6; ++j) {
    int col = j * 16 + (lane & 15);
#pragma unroll
    for (int r = 0; r < 8; ++r)
      cat[(size_t)(m0 + r + (hi << 3)) * 384 + coff + col] = (_Float16)acc[j][r];
  }
}

// fuse: 384 -> 96 (W staged whole, 72 KB), f32 out for instance-norm stats
__global__ __launch_bounds__(256) void k_gemm_fuse(
    const _Float16* __restrict__ cat, const _Float16* __restrict__ wt,
    float* __restrict__ fo) {
  __shared__ __align__(16) _Float16 wslab[96 * 384];
  int tid = threadIdx.x;
  int lane = tid & 31, wave = tid >> 5, hi = lane >> 4;
  int m0 = blockIdx.x * 128 + wave * 16;
  stage_w(wslab, wt, 96 * 384, 384, 96, tid);
  const _Float16* row = cat + (size_t)(m0 + (lane & 15)) * 384;
  tdm_wait();
  __syncthreads();
  v8f acc[6] = {};
#pragma unroll
  for (int k0 = 0; k0 < 384; k0 += 32) {
    v16h a = a_frag(row, k0, hi);
#pragma unroll
    for (int j = 0; j < 6; ++j)
      acc[j] = wmma32(a, b_frag_wt(wslab, 384, j * 16, k0, lane), acc[j]);
  }
#pragma unroll
  for (int j = 0; j < 6; ++j) {
    int col = j * 16 + (lane & 15);
#pragma unroll
    for (int r = 0; r < 8; ++r)
      fo[(size_t)(m0 + r + (hi << 3)) * 96 + col] = acc[j][r];
  }
}

// ------------------------------- host orchestration ------------------------
// Input flattening (jax pytree, dict keys sorted):
//  per block (stride 65): b1,b2,b3,b4, 4x branch{fc1_b,fc1_w,fc2_b,fc2_w,
//  n1_b,n1_g,n2_b,n2_g,proj_b,proj_w,qkv_b,qkv_w,rpb}, fuse,gate_b1,gate_b2,
//  gate_w1,gate_w2,in_b,in_g,ls_a,ls_c; then final_ls(130), x(131).

extern "C" void kernel_launch(void* const* d_in, const int* in_sizes, int n_in,
                              void* d_out, int out_size, void* d_ws, size_t ws_size,
                              hipStream_t stream) {
  (void)in_sizes; (void)n_in; (void)out_size; (void)ws_size;
  const int WSH[4][3] = {{4, 4, 4}, {2, 8, 8}, {8, 2, 8}, {8, 8, 2}};
  auto F = [&](int i) -> const float* { return (const float*)d_in[i]; };
  const float* xin = F(131);

  char* ws = (char*)d_ws;
  size_t off = 0;
  auto take = [&](size_t bytes) -> void* {
    void* p = ws + off;
    off = (off + bytes + 255) & ~(size_t)255;
    return p;
  };
  float*    xcl    = (float*)take((size_t)V_ * 96 * 4);
  float*    ybuf   = (float*)take((size_t)V_ * 96 * 4);
  float*    xafter = (float*)take((size_t)V_ * 96 * 4);
  float*    fuseo  = (float*)take((size_t)V_ * 96 * 4);
  _Float16* xln    = (_Float16*)take((size_t)V_ * 96 * 2);
  _Float16* aout   = (_Float16*)take((size_t)V_ * 96 * 2);
  _Float16* xh16   = (_Float16*)take((size_t)V_ * 96 * 2);
  _Float16* qkvb   = (_Float16*)take((size_t)V_ * 384 * 2);  // head-padded q,k,v
  _Float16* b384   = (_Float16*)take((size_t)V_ * 384 * 2);  // MLP hidden / conv cat
  _Float16* zrow   = (_Float16*)take(256);
  float*    gm     = (float*)take(96 * 4);
  float*    gwv    = (float*)take(64);
  float*    meanb  = (float*)take(96 * 4);
  float*    varb   = (float*)take(96 * 4);

  _Float16 *wq[2][4], *wp[2][4], *w1[2][4], *w2[2][4];
  _Float16 *b1w[2], *b3w[2], *b4w[2], *fw[2];
  auto cvt = [&](const float* src, size_t n) -> _Float16* {
    _Float16* d = (_Float16*)take(n * 2);
    k_cvt<<<dim3((unsigned)((n + 255) / 256)), 256, 0, stream>>>(src, d, (int)n);
    return d;
  };
  for (int blk = 0; blk < 2; ++blk) {
    int base = blk * 65;
    for (int br = 0; br < 4; ++br) {
      int bb = base + 4 + br * 13;
      w1[blk][br] = cvt(F(bb + 1), 384 * 96);
      w2[blk][br] = cvt(F(bb + 3), 96 * 384);
      wp[blk][br] = cvt(F(bb + 9), 96 * 96);
      wq[blk][br] = cvt(F(bb + 11), 288 * 96);
    }
    b1w[blk] = (_Float16*)take((size_t)25 * 9216 * 2);
    k_cvt_convw<<<dim3((25 * 9216 + 255) / 256), 256, 0, stream>>>(F(base + 0), b1w[blk], 25);
    b3w[blk] = (_Float16*)take((size_t)5 * 9216 * 2);
    k_cvt_convw<<<dim3((5 * 9216 + 255) / 256), 256, 0, stream>>>(F(base + 2), b3w[blk], 5);
    b4w[blk] = (_Float16*)take((size_t)5 * 9216 * 2);
    k_cvt_convw<<<dim3((5 * 9216 + 255) / 256), 256, 0, stream>>>(F(base + 3), b4w[blk], 5);
    fw[blk] = cvt(F(base + 56), 96 * 384);
  }
  k_fill0h<<<dim3((V_ * 384) / 256), 256, 0, stream>>>(qkvb, (size_t)V_ * 384);
  k_fill0h<<<1, 128, 0, stream>>>(zrow, 128);
  k_init_xcl<<<dim3((96 * V_) / 256), 256, 0, stream>>>(xin, xcl);

  for (int blk = 0; blk < 2; ++blk) {
    int base = blk * 65;
    k_chan_mean<<<96, 256, 0, stream>>>(xcl, gm);
    k_gate<<<1, 32, 0, stream>>>(gm, F(base + 59), F(base + 57), F(base + 60),
                                 F(base + 58), gwv);
    for (int br = 0; br < 4; ++br) {
      int bb = base + 4 + br * 13;
      int wd = WSH[br][0], wh = WSH[br][1], ww = WSH[br][2];
      int N = wd * wh * ww;
      int shifted = (blk & 1);
      int sd = shifted ? wd / 2 : 0, sh = shifted ? wh / 2 : 0, sw = shifted ? ww / 2 : 0;
      k_ln<<<128, 256, 0, stream>>>(xcl, F(bb + 5), F(bb + 4), xln);
      k_gemm_qkv<<<dim3(256, 3), 256, 0, stream>>>(xln, wq[blk][br], F(bb + 10),
                                                   qkvb, wd, wh, ww, sd, sh, sw);
      if (N == 64)
        k_attn<4><<<dim3(512, 4), 128, 0, stream>>>(qkvb, F(bb + 12), aout,
                                                    wd, wh, ww, shifted);
      else
        k_attn<8><<<dim3(256, 4), 256, 0, stream>>>(qkvb, F(bb + 12), aout,
                                                    wd, wh, ww, shifted);
      k_gemm_proj<<<dim3(256, 1), 256, 0, stream>>>(aout, wp[blk][br], F(bb + 8),
                                                    xcl, xafter, wd, wh, ww, sd, sh, sw);
      k_ln<<<128, 256, 0, stream>>>(xafter, F(bb + 7), F(bb + 6), xln);
      k_gemm_fc1<<<dim3(256, 4), 256, 0, stream>>>(xln, w1[blk][br], F(bb + 0), b384);
      k_gemm_fc2<<<dim3(256, 1), 256, 0, stream>>>(b384, w2[blk][br], F(bb + 2),
                                                   xafter, ybuf, gwv, br);
    }
    k_xupd<<<dim3((V_ * 96) / 256), 256, 0, stream>>>(xcl, ybuf, F(base + 63), xh16);
    k_gemm_conv<<<dim3(256, 1), 256, 0, stream>>>(xh16, b1w[blk], b384, zrow, 5, 5, 2, 2, 0);
    k_dwconv<<<dim3((V_ * 96) / 256), 256, 0, stream>>>(xcl, F(base + 1), b384);
    k_gemm_conv<<<dim3(256, 1), 256, 0, stream>>>(xh16, b3w[blk], b384, zrow, 1, 5, 0, 2, 192);
    k_gemm_conv<<<dim3(256, 1), 256, 0, stream>>>(xh16, b4w[blk], b384, zrow, 5, 1, 2, 0, 288);
    k_gemm_fuse<<<dim3(256, 1), 256, 0, stream>>>(b384, fw[blk], fuseo);
    k_in_stats<<<96, 256, 0, stream>>>(fuseo, meanb, varb);
    k_in_apply<<<dim3((V_ * 96) / 256), 256, 0, stream>>>(xcl, fuseo, meanb, varb,
                                                          F(base + 62), F(base + 61),
                                                          F(base + 64));
  }
  k_final<<<dim3((96 * V_) / 256), 256, 0, stream>>>(xcl, xin, F(130), (float*)d_out);
}
